// BSRBF_KANLayer_17471926960416
// MI455X (gfx1250) — matrix operand order repaired and checked by reading, not compiled, not run on hardware
//
#include <hip/hip_runtime.h>

typedef __bf16 bf16;
typedef bf16 v16bf __attribute__((ext_vector_type(16)));
typedef float v8f __attribute__((ext_vector_type(8)));
typedef int v4i __attribute__((vector_size(16)));
typedef __attribute__((address_space(1))) v4i as1_v4i;
typedef __attribute__((address_space(3))) v4i as3_v4i;

#define D_IN   512
#define D_OUT  512
#define NTOK   16384      // B*S
#define MT     64         // tokens per block
#define KC     64         // K chunk
#define NCHUNK 72         // 8 base (512) + 64 spline (4096)
#define APITCH 72         // padded bf16 row pitch (144B = 36 dwords, conflict-free)
#define BPITCH 72
#define XNP    516        // f32 pitch for xn rows

// ws: Wt[72][512][72] bf16, chunked + padded, written by pre-pass
#define WT_CHUNK_ELEMS (D_OUT * BPITCH)       // 36864 bf16
#define SEGS_PER_CHUNK (WT_CHUNK_ELEMS / 8)   // 4608 x 16B

// LDS layout (bytes)
#define XS_BYTES  (MT * XNP * 4)              // 132096
#define AT_BYTES  (MT * APITCH * 2)           // 9216
#define BT_BYTES  (D_OUT * BPITCH * 2)        // 73728
#define AT0_OFF   (XS_BYTES)
#define BT0_OFF   (AT0_OFF + 2 * AT_BYTES)
#define SMEM_BYTES (BT0_OFF + 2 * BT_BYTES)   // 297984 < 320KB

union Pack8 { bf16 h[8]; uint4 q; };
union Frag  { v16bf v; uint4 q[2]; };

__device__ __forceinline__ void async_copy16(const void* g, void* l) {
#if defined(__has_builtin) && __has_builtin(__builtin_amdgcn_global_load_async_to_lds_b128)
  __builtin_amdgcn_global_load_async_to_lds_b128(
      (as1_v4i*)(g), (as3_v4i*)(l), 0, 0);
#else
  *(uint4*)l = *(const uint4*)g;
#endif
}

__device__ __forceinline__ void wait_async() {
#if defined(__has_builtin) && __has_builtin(__builtin_amdgcn_s_wait_asynccnt)
  __builtin_amdgcn_s_wait_asynccnt(0);
#else
  asm volatile("s_wait_asynccnt 0" ::: "memory");
#endif
}

// ---------------- Pre-pass: f32 weights -> chunked/padded bf16 in ws ----------------
__global__ __launch_bounds__(256) void kan_wprep(const float* __restrict__ Wb,
                                                 const float* __restrict__ Ws,
                                                 bf16* __restrict__ Wt) {
  const int flat = blockIdx.x * 256 + threadIdx.x;  // [0, 72*512*9)
  const int c    = flat / (D_OUT * 9);
  const int rem  = flat - c * (D_OUT * 9);
  const int row  = rem / 9;
  const int seg  = rem - row * 9;
  Pack8 pk;
  if (seg == 8) {
#pragma unroll
    for (int j = 0; j < 8; ++j) pk.h[j] = (bf16)0.f;   // pad lane
  } else {
    const float* src = (c < 8)
        ? (Wb + (size_t)row * D_IN + c * KC + seg * 8)
        : (Ws + (size_t)row * (D_IN * 8) + (c - 8) * KC + seg * 8);
#pragma unroll
    for (int j = 0; j < 8; ++j) pk.h[j] = (bf16)src[j];
  }
  *(uint4*)(Wt + (size_t)c * WT_CHUNK_ELEMS + row * BPITCH + seg * 8) = pk.q;
}

// ---------------- Fused LN + activations + GEMM ----------------
__global__ __launch_bounds__(256) void kan_fused(
    const float* __restrict__ x, const float* __restrict__ gamma,
    const float* __restrict__ beta, const bf16* __restrict__ Wt,
    float* __restrict__ out)
{
  extern __shared__ char smem[];
  float* xs = (float*)smem;

  const int tid    = threadIdx.x;
  const int lane   = tid & 31;
  const int wave   = tid >> 5;
  const int wave_m = wave & 1;   // 2 x 32 rows
  const int wave_n = wave >> 1;  // 4 x 128 cols
  const int row0   = blockIdx.x * MT;

  // ---- Phase 1: LayerNorm -> LDS (f32) ----
  for (int t = wave * 8; t < wave * 8 + 8; ++t) {
    const float* xr = x + (size_t)(row0 + t) * D_IN;
    float v[16], s = 0.f, ss = 0.f;
#pragma unroll
    for (int i = 0; i < 16; ++i) {
      float xv = xr[lane + 32 * i];
      v[i] = xv; s += xv; ss += xv * xv;
    }
#pragma unroll
    for (int o = 16; o >= 1; o >>= 1) {
      s  += __shfl_xor(s,  o, 32);
      ss += __shfl_xor(ss, o, 32);
    }
    const float mu  = s * (1.f / D_IN);
    const float var = ss * (1.f / D_IN) - mu * mu;
    const float inv = rsqrtf(var + 1e-5f);
#pragma unroll
    for (int i = 0; i < 16; ++i) {
      const int d = lane + 32 * i;
      xs[t * XNP + d] = (v[i] - mu) * inv * gamma[d] + beta[d];
    }
  }

  v8f acc[2][8];
#pragma unroll
  for (int mt = 0; mt < 2; ++mt)
#pragma unroll
    for (int nt = 0; nt < 8; ++nt) {
      v8f z = {0.f, 0.f, 0.f, 0.f, 0.f, 0.f, 0.f, 0.f};
      acc[mt][nt] = z;
    }

  // ---- staging helper ----
  auto stage = [&](int c, char* At_b, char* Bt_b) {
    // B: async identity copy of pre-chunked bf16 weights (4608 x 16B)
    {
      const char* gsrc = (const char*)(Wt + (size_t)c * WT_CHUNK_ELEMS);
#pragma unroll
      for (int i = 0; i < SEGS_PER_CHUNK / 256; ++i) {  // 18
        const int s = i * 256 + tid;
        async_copy16(gsrc + (size_t)s * 16, Bt_b + (size_t)s * 16);
      }
    }
    // A: features
    bf16* At = (bf16*)At_b;
    if (c < 8) {
      const int t  = tid >> 2;
      const int i0 = (tid & 3) * 16;
      const float* src = &xs[t * XNP + c * KC + i0];
      Pack8 p0, p1;
#pragma unroll
      for (int j = 0; j < 8; ++j) {
        float a = src[j], b = src[j + 8];
        p0.h[j] = (bf16)(a > 0.f ? a : 0.f);
        p1.h[j] = (bf16)(b > 0.f ? b : 0.f);
      }
      *(uint4*)(&At[t * APITCH + i0])     = p0.q;
      *(uint4*)(&At[t * APITCH + i0 + 8]) = p1.q;
    } else {
      const int cs = c - 8;
#pragma unroll
      for (int p = tid; p < MT * 8; p += 256) {
        const int t  = p >> 3;
        const int di = p & 7;
        const float xv = xs[t * XNP + cs * 8 + di];
        // cubic B-spline via truncated powers:
        // b_j = (T_j - 4T_{j+1} + 6T_{j+2} - 4T_{j+3} + T_{j+4}) / 6,  T_k = max(S-k,0)^3
        const float S = (xv + 3.3f) * (1.f / 0.6f);
        float T[12];
#pragma unroll
        for (int k = 0; k < 12; ++k) {
          float d = fmaxf(S - (float)k, 0.f);
          T[k] = d * d * d;
        }
        const float E = xv * (7.f / 3.f) + 3.5f;  // (xv - g_j)/denom = E - j
        Pack8 pk;
#pragma unroll
        for (int j = 0; j < 8; ++j) {
          float t1 = T[j] + T[j + 4];
          float t2 = T[j + 1] + T[j + 3];
          float bs = (t1 + 6.f * T[j + 2] - 4.f * t2) * (1.f / 6.f);
          float e  = E - (float)j;
          pk.h[j]  = (bf16)(bs + __expf(-e * e));
        }
        *(uint4*)(&At[t * APITCH + di * 8]) = pk.q;
      }
    }
  };

  auto compute = [&](const char* At_b, const char* Bt_b) {
    const bf16* At = (const bf16*)At_b;
    const bf16* Bt = (const bf16*)Bt_b;
    const int rsel = lane & 15;
#pragma unroll
    for (int ks = 0; ks < 2; ++ks) {
      const int akb = ks * 32 + ((lane & 16) ? 8 : 0);
      const int bkb = ks * 32 + ((lane & 16) ? 8 : 0);
      Frag a0, a1;
      {
        const bf16* p = &At[(wave_m * 32 + rsel) * APITCH + akb];
        a0.q[0] = *(const uint4*)p;       a0.q[1] = *(const uint4*)(p + 16);
        p += 16 * APITCH;
        a1.q[0] = *(const uint4*)p;       a1.q[1] = *(const uint4*)(p + 16);
      }
#pragma unroll
      for (int nt = 0; nt < 8; ++nt) {
        Frag b;
        const bf16* p = &Bt[(wave_n * 128 + nt * 16 + rsel) * BPITCH + bkb];
        b.q[0] = *(const uint4*)p;        b.q[1] = *(const uint4*)(p + 16);
        acc[0][nt] = __builtin_amdgcn_wmma_f32_16x16x32_bf16(
            false, a0.v, false, b.v, (short)0, acc[0][nt], false, false);
        acc[1][nt] = __builtin_amdgcn_wmma_f32_16x16x32_bf16(
            false, a1.v, false, b.v, (short)0, acc[1][nt], false, false);
      }
    }
  };

  __syncthreads();                 // xn visible
  stage(0, smem + AT0_OFF, smem + BT0_OFF);
  wait_async();
  __syncthreads();

  for (int c = 0; c < NCHUNK; ++c) {
    if (c + 1 < NCHUNK) {
      const int nx = (c + 1) & 1;
      stage(c + 1, smem + AT0_OFF + nx * AT_BYTES,
                   smem + BT0_OFF + nx * BT_BYTES);
    }
    const int cu = c & 1;
    compute(smem + AT0_OFF + cu * AT_BYTES,
            smem + BT0_OFF + cu * BT_BYTES);
    wait_async();
    __syncthreads();
  }

  // ---- Epilogue ----
#pragma unroll
  for (int mt = 0; mt < 2; ++mt) {
    const int mrow = row0 + wave_m * 32 + mt * 16 + ((lane >> 4) << 3);
#pragma unroll
    for (int nt = 0; nt < 8; ++nt) {
      const int ncol = wave_n * 128 + nt * 16 + (lane & 15);
#pragma unroll
      for (int r = 0; r < 8; ++r)
        out[(size_t)(mrow + r) * D_OUT + ncol] = acc[mt][nt][r];
    }
  }
}

extern "C" void kernel_launch(void* const* d_in, const int* in_sizes, int n_in,
                              void* d_out, int out_size, void* d_ws, size_t ws_size,
                              hipStream_t stream) {
  (void)in_sizes; (void)n_in; (void)out_size; (void)ws_size;
  const float* x     = (const float*)d_in[0];
  const float* gamma = (const float*)d_in[1];
  const float* beta  = (const float*)d_in[2];
  const float* Wb    = (const float*)d_in[3];
  const float* Ws    = (const float*)d_in[4];
  float* out = (float*)d_out;
  bf16* Wt   = (bf16*)d_ws;   // needs 72*512*72*2 = 5.3 MB

  // Pre-pass: 72*512*9 segments, one per thread
  kan_wprep<<<dim3((NCHUNK * D_OUT * 9) / 256, 1, 1), 256, 0, stream>>>(Wb, Ws, Wt);

  (void)hipFuncSetAttribute((const void*)kan_fused,
                            hipFuncAttributeMaxDynamicSharedMemorySize,
                            (int)SMEM_BYTES);
  kan_fused<<<dim3(NTOK / MT, 1, 1), 256, SMEM_BYTES, stream>>>(
      x, gamma, beta, Wt, out);
}
